// TrackMamba_66073776881828
// MI455X (gfx1250) — matrix operand order repaired and checked by reading, not compiled, not run on hardware
//
#include <hip/hip_runtime.h>

// ---------------- problem constants (match reference) ----------------
constexpr int B_ = 4;
constexpr int L_ = 4096;
constexpr int D_ = 256;
constexpr int BL = B_ * L_;                    // 16384
constexpr int DSTATE = 64;
constexpr int HEADDIM = 64;
constexpr int DIN = 2 * D_;                    // 512
constexpr int NHEADS = DIN / HEADDIM;          // 8
constexpr int CONVDIM = DIN + 2 * DSTATE;      // 640
constexpr int DIP = 2 * DIN + 2 * DSTATE + NHEADS; // 1160
constexpr int DCONV = 4;
constexpr int NLAYERS = 4;
constexpr int CHUNK = 64;
constexpr int NC = L_ / CHUNK;                 // 64
constexpr float EPS_ = 1e-5f;

typedef __attribute__((ext_vector_type(16))) _Float16 v16h;
typedef __attribute__((ext_vector_type(8)))  float    v8f;

typedef __attribute__((ext_vector_type(4))) unsigned int u32x4;
typedef __attribute__((ext_vector_type(8))) int          i32x8;
typedef __attribute__((ext_vector_type(4))) int          i32x4;

#if __has_builtin(__builtin_amdgcn_tensor_load_to_lds)
#define USE_TDM 1
#else
#define USE_TDM 0
#endif

// =====================================================================
// TDM helper: DMA a 64x64 f32 tile (row stride = stride_elems) from
// global memory into LDS with a 1-dword pad every 64 dwords, giving the
// LDS tile a 65-float row pitch (bank-conflict-free A-side reads).
// D# per CDNA5 ISA ch.8: group0 {count|flags, lds_addr, gaddr lo,
// gaddr[56:32]|type=2}; group1 {dsize/pad flags, dims, tile dims, stride}.
// =====================================================================
#if USE_TDM
__device__ __forceinline__ void tdm_load_tile_f32_64x64(const float* gsrc,
                                                        const float* lds_dst,
                                                        unsigned stride_elems) {
  unsigned long long ga = (unsigned long long)(uintptr_t)gsrc;
  unsigned lds_off = (unsigned)(uintptr_t)lds_dst;   // generic LDS ptr: low 32b = LDS offset
  u32x4 g0;
  g0.x = (unsigned)__builtin_amdgcn_readfirstlane(1);               // count=1
  g0.y = (unsigned)__builtin_amdgcn_readfirstlane((int)lds_off);
  g0.z = (unsigned)__builtin_amdgcn_readfirstlane((int)(unsigned)(ga & 0xffffffffull));
  g0.w = (unsigned)__builtin_amdgcn_readfirstlane(
      (int)(unsigned)(((ga >> 32) & 0x01ffffffull) | (2u << 30)));  // addr[56:32] | type=2
  i32x8 g1;
  // data_size=2 (4B) @ [17:16]; pad_enable @ [20]; pad_interval=5 (64 dw) @ [24:22];
  // pad_amount=0 (1 dw) @ [31:25]; wg_mask=0 (not in cluster)
  g1[0] = __builtin_amdgcn_readfirstlane((int)((2u << 16) | (1u << 20) | (5u << 22)));
  g1[1] = __builtin_amdgcn_readfirstlane((int)(64u << 16));   // tensor_dim0[15:0] @ bits[63:48]
  g1[2] = __builtin_amdgcn_readfirstlane((int)(64u << 16));   // tensor_dim1[15:0] @ bits[95:80]
  g1[3] = __builtin_amdgcn_readfirstlane((int)(64u << 16));   // tile_dim0 @ bits[127:112]
  g1[4] = __builtin_amdgcn_readfirstlane((int)64);            // tile_dim1=64, tile_dim2=0
  g1[5] = __builtin_amdgcn_readfirstlane((int)stride_elems);  // tensor_dim0_stride[31:0]
  g1[6] = 0;
  g1[7] = 0;
  i32x4 gz = {0, 0, 0, 0};
#if defined(__clang_major__) && (__clang_major__ >= 23)
  i32x8 gz8 = {0, 0, 0, 0, 0, 0, 0, 0};
  __builtin_amdgcn_tensor_load_to_lds(g0, g1, gz, gz, gz8, 0);
#else
  __builtin_amdgcn_tensor_load_to_lds(g0, g1, gz, gz, 0);
#endif
}
#endif

// Fallback cooperative staging (256 threads) into the padded LDS tile.
__device__ __forceinline__ void stage_tile_f32(float (*dst)[65],
                                               const float* __restrict__ g,
                                               int stride, int tid) {
  for (int e = tid; e < 64 * 16; e += 256) {
    int r = e >> 4;
    int c4 = (e & 15) << 2;
    const float* src = g + (size_t)r * stride + c4;
    dst[r][c4 + 0] = src[0];
    dst[r][c4 + 1] = src[1];
    dst[r][c4 + 2] = src[2];
    dst[r][c4 + 3] = src[3];
  }
}

// =====================================================================
// 0) Embedding: h[m,d] = emb[x[m],d] + pos[x[m],d]
// =====================================================================
__global__ void embed_kernel(const int* __restrict__ x,
                             const float* __restrict__ emb,
                             const float* __restrict__ pos,
                             float* __restrict__ h) {
  int idx = blockIdx.x * blockDim.x + threadIdx.x;
  if (idx >= BL * D_) return;
  int d = idx % D_;
  int m = idx / D_;
  int tok = x[m];
  h[idx] = emb[tok * D_ + d] + pos[tok * D_ + d];
}

// =====================================================================
// 1) Generic NT GEMM with WMMA: C[M,N] = A[M,K] * W[N,K]^T  (f32 in/out,
//    f16 multiplicands, f32 accumulate). One 16x16 tile per wave.
// =====================================================================
__global__ void gemm_nt_wmma(const float* __restrict__ A,
                             const float* __restrict__ W,
                             float* __restrict__ Cout,
                             int M, int N, int K) {
  int gtid = blockIdx.x * blockDim.x + threadIdx.x;
  int wave = gtid >> 5;
  int lane = gtid & 31;
  int ntn = (N + 15) >> 4;
  int nwaves = (M >> 4) * ntn;
  if (wave >= nwaves) return;                 // wave-uniform exit (EXEC all-1 for WMMA)
  int tm = wave / ntn, tn = wave % ntn;
  int lr = lane & 15;
  int hi = lane >> 4;
  int kbA = hi ? 8 : 0;                       // A: K half-offset per ISA layout
  int kbB = hi ? 8 : 0;                       // B: K half-offset per ISA layout
  int arow = tm * 16 + lr;
  int bcol = tn * 16 + lr;
  const float* aptr = A + (size_t)arow * K;
  const float* wptr = W + (size_t)bcol * K;   // only dereferenced when bcol < N
  bool bok = (bcol < N);

  v8f acc = {};
  for (int k0 = 0; k0 < K; k0 += 32) {
    __builtin_prefetch(aptr + k0 + 128, 0, 1); // gfx1250 global_prefetch_b8
    v16h af, bf;
#pragma unroll
    for (int i = 0; i < 8; ++i) {
      af[i]     = (_Float16)aptr[k0 + kbA + i];
      af[8 + i] = (_Float16)aptr[k0 + 16 + kbA + i];
    }
    if (bok) {
#pragma unroll
      for (int j = 0; j < 16; ++j) bf[j] = (_Float16)wptr[k0 + kbB + (j & 7) + ((j >> 3) << 4)];
    } else {
#pragma unroll
      for (int j = 0; j < 16; ++j) bf[j] = (_Float16)0.0f;
    }
    acc = __builtin_amdgcn_wmma_f32_16x16x32_f16(false, af, false, bf,
                                                 (short)0, acc, false, false);
  }
  int orow = tm * 16 + hi * 8;
  int ocol = tn * 16 + lr;
  if (ocol < N) {
#pragma unroll
    for (int r = 0; r < 8; ++r)
      Cout[(size_t)(orow + r) * N + ocol] = acc[r];
  }
}

// =====================================================================
// 2) Depthwise causal conv (width 4) + bias + SiLU over xBC channels
// =====================================================================
__global__ void conv_silu_kernel(const float* __restrict__ zxbcdt,
                                 const float* __restrict__ cw,
                                 const float* __restrict__ cb,
                                 float* __restrict__ xBCc) {
  int idx = blockIdx.x * blockDim.x + threadIdx.x;
  if (idx >= BL * CONVDIM) return;
  int cch = idx % CONVDIM;
  int m = idx / CONVDIM;
  int l = m % L_;
  float acc = cb[cch];
#pragma unroll
  for (int j = 0; j < DCONV; ++j) {
    int li = l - (DCONV - 1) + j;
    if (li >= 0)
      acc += cw[cch * DCONV + j] *
             zxbcdt[(size_t)(m - (DCONV - 1) + j) * DIP + DIN + cch];
  }
  xBCc[(size_t)m * CONVDIM + cch] = acc / (1.f + __expf(-acc));
}

// =====================================================================
// 3) dt = softplus(dt_raw + dt_bias)
// =====================================================================
__global__ void dt_softplus_kernel(const float* __restrict__ zxbcdt,
                                   const float* __restrict__ dtb,
                                   float* __restrict__ dt_sp) {
  int idx = blockIdx.x * blockDim.x + threadIdx.x;
  if (idx >= BL * NHEADS) return;
  int h = idx % NHEADS;
  int m = idx / NHEADS;
  float xv = zxbcdt[(size_t)m * DIP + DIN + CONVDIM + h] + dtb[h];
  dt_sp[idx] = (xv > 20.f) ? xv : log1pf(__expf(xv));
}

// =====================================================================
// 4) per-chunk inclusive log-cumsum of dt*A  (A = -exp(A_log))
// =====================================================================
__global__ void clsum_kernel(const float* __restrict__ dt_sp,
                             const float* __restrict__ Alog,
                             float* __restrict__ clsum) {
  int idx = blockIdx.x * blockDim.x + threadIdx.x;
  if (idx >= B_ * NHEADS * NC) return;
  int c = idx % NC;
  int bh = idx / NC;
  int h = bh % NHEADS;
  int b = bh / NHEADS;
  float Ah = -__expf(Alog[h]);
  size_t rowbase = (size_t)b * L_ + c * CHUNK;
  float acc = 0.f;
  for (int t = 0; t < CHUNK; ++t) {
    acc += dt_sp[(rowbase + t) * NHEADS + h] * Ah;
    clsum[(rowbase + t) * NHEADS + h] = acc;
  }
}

// =====================================================================
// 5) Chunk-state GEMM (WMMA): Sc[p,n] = sum_t xh[t,p] * (dt[t]*decay[t]*B[t,n]).
//    xh/B 64x64 tiles staged into LDS via TDM (pad pitch 65).
// =====================================================================
__global__ void chunk_state_wmma(const float* __restrict__ xBCc,
                                 const float* __restrict__ dt_sp,
                                 const float* __restrict__ clsum,
                                 float* __restrict__ Sc) {
  int blk = blockIdx.x;
  int c = blk % NC;
  int bh = blk / NC;
  int h = bh % NHEADS;
  int b = bh / NHEADS;
  int tid = threadIdx.x;
  int lane = tid & 31, wv = tid >> 5;
  __shared__ float XH[CHUNK][65];
  __shared__ float BT[CHUNK][65];
  __shared__ float wk[CHUNK];
  size_t rowbase = (size_t)b * L_ + c * CHUNK;
  if (tid < CHUNK) {
    float cl_t = clsum[(rowbase + tid) * NHEADS + h];
    float cl_last = clsum[(rowbase + CHUNK - 1) * NHEADS + h];
    float dtv = dt_sp[(rowbase + tid) * NHEADS + h];
    wk[tid] = dtv * __expf(cl_last - cl_t);
  }
  const float* gxh = xBCc + rowbase * CONVDIM + h * HEADDIM;
  const float* gB  = xBCc + rowbase * CONVDIM + DIN;
#if USE_TDM
  if (wv == 0) {
    tdm_load_tile_f32_64x64(gxh, &XH[0][0], CONVDIM);
    tdm_load_tile_f32_64x64(gB,  &BT[0][0], CONVDIM);
    __builtin_amdgcn_s_wait_tensorcnt(0);
  }
#else
  stage_tile_f32(XH, gxh, CONVDIM, tid);
  stage_tile_f32(BT, gB, CONVDIM, tid);
#endif
  __syncthreads();

  int lr = lane & 15, hi = lane >> 4;
  int kbA = hi ? 8 : 0, kbB = hi ? 8 : 0;
  for (int q = wv; q < 16; q += 8) {          // 2 tiles per wave (uniform)
    int tp = q >> 2, tn = q & 3;
    int p = tp * 16 + lr;                     // A row index (head dim)
    int ncol = tn * 16 + lr;                  // B col index (state dim)
    v8f acc = {};
    for (int k0 = 0; k0 < CHUNK; k0 += 32) {  // K = t
      v16h af, bf;
#pragma unroll
      for (int i = 0; i < 8; ++i) {
        af[i]     = (_Float16)XH[k0 + kbA + i][p];
        af[8 + i] = (_Float16)XH[k0 + 16 + kbA + i][p];
      }
#pragma unroll
      for (int j = 0; j < 16; ++j) {
        int t = k0 + kbB + (j & 7) + ((j >> 3) << 4);
        bf[j] = (_Float16)(BT[t][ncol] * wk[t]);
      }
      acc = __builtin_amdgcn_wmma_f32_16x16x32_f16(false, af, false, bf,
                                                   (short)0, acc, false, false);
    }
    int prow = tp * 16 + hi * 8;
    float* out = Sc + ((size_t)bh * NC + c) * (HEADDIM * DSTATE);
#pragma unroll
    for (int r = 0; r < 8; ++r)
      out[(prow + r) * DSTATE + tn * 16 + lr] = acc[r];
  }
}

// =====================================================================
// 6) Sequential inter-chunk recurrence (64 steps only). In-place safe
//    when hstart == Sc (read-before-write per element).
// =====================================================================
__global__ void state_scan_kernel(const float* Sc, const float* clsum,
                                  float* hstart) {
  int bh = blockIdx.x;                        // B*NHEADS
  int h = bh % NHEADS;
  int b = bh / NHEADS;
  int tid = threadIdx.x;                      // 256 threads x 16 elems = 4096
  float s[16];
#pragma unroll
  for (int i = 0; i < 16; ++i) s[i] = 0.f;
  int e0 = tid * 16;
  for (int c = 0; c < NC; ++c) {
    size_t base = ((size_t)bh * NC + c) * (HEADDIM * DSTATE) + e0;
    float P = __expf(clsum[((size_t)b * L_ + c * CHUNK + CHUNK - 1) * NHEADS + h]);
#pragma unroll
    for (int i = 0; i < 16; ++i) {
      float sc = Sc[base + i];
      hstart[base + i] = s[i];
      s[i] = s[i] * P + sc;
    }
  }
}

// =====================================================================
// 7) y-chunk kernel (WMMA x3): G = tril(C B^T) .* exp(cls_t - cls_s) (LDS),
//    Y = G * Xdt  +  diag(exp(cls_t)) * C * hstart^T, then + D_skip*xh.
//    xh/B/C tiles staged via TDM.
// =====================================================================
__global__ void y_chunk_wmma(const float* __restrict__ xBCc,
                             const float* __restrict__ dt_sp,
                             const float* __restrict__ clsum,
                             const float* __restrict__ hstart,
                             const float* __restrict__ Dskip,
                             float* __restrict__ ys) {
  int blk = blockIdx.x;
  int c = blk % NC;
  int bh = blk / NC;
  int h = bh % NHEADS;
  int b = bh / NHEADS;
  int tid = threadIdx.x, lane = tid & 31, wv = tid >> 5;
  __shared__ float XH[CHUNK][65];
  __shared__ float BM[CHUNK][65];
  __shared__ float CM[CHUNK][65];
  __shared__ _Float16 G[CHUNK][68];
  __shared__ float cls[CHUNK];
  __shared__ float dts[CHUNK];
  size_t rowbase = (size_t)b * L_ + c * CHUNK;
  if (tid < CHUNK) {
    cls[tid] = clsum[(rowbase + tid) * NHEADS + h];
    dts[tid] = dt_sp[(rowbase + tid) * NHEADS + h];
  }
  const float* gxh = xBCc + rowbase * CONVDIM + h * HEADDIM;
  const float* gB  = xBCc + rowbase * CONVDIM + DIN;
  const float* gC  = xBCc + rowbase * CONVDIM + DIN + DSTATE;
#if USE_TDM
  if (wv == 0) {
    tdm_load_tile_f32_64x64(gxh, &XH[0][0], CONVDIM);
    tdm_load_tile_f32_64x64(gB,  &BM[0][0], CONVDIM);
    tdm_load_tile_f32_64x64(gC,  &CM[0][0], CONVDIM);
    __builtin_amdgcn_s_wait_tensorcnt(0);
  }
#else
  stage_tile_f32(XH, gxh, CONVDIM, tid);
  stage_tile_f32(BM, gB, CONVDIM, tid);
  stage_tile_f32(CM, gC, CONVDIM, tid);
#endif
  __syncthreads();

  int lr = lane & 15, hi = lane >> 4;
  int kbA = hi ? 8 : 0, kbB = hi ? 8 : 0;

  // ---- Phase 1: G[t][s] = tril(C B^T) .* exp(cls_t - cls_s) ----
  for (int q = wv; q < 16; q += 8) {
    int tt = q >> 2, tsd = q & 3;
    int trow = tt * 16 + lr;                  // A row (t)
    int scol = tsd * 16 + lr;                 // B col (s)
    v8f acc = {};
    for (int k0 = 0; k0 < DSTATE; k0 += 32) { // K = n
      v16h af, bf;
#pragma unroll
      for (int i = 0; i < 8; ++i) {
        af[i]     = (_Float16)CM[trow][k0 + kbA + i];
        af[8 + i] = (_Float16)CM[trow][k0 + 16 + kbA + i];
      }
#pragma unroll
      for (int j = 0; j < 16; ++j) bf[j] = (_Float16)BM[scol][k0 + kbB + (j & 7) + ((j >> 3) << 4)];
      acc = __builtin_amdgcn_wmma_f32_16x16x32_f16(false, af, false, bf,
                                                   (short)0, acc, false, false);
    }
    int t0 = tt * 16 + hi * 8;
    int sI = tsd * 16 + lr;
    float clsS = cls[sI];
#pragma unroll
    for (int r = 0; r < 8; ++r) {
      int t = t0 + r;
      float g = (sI <= t) ? acc[r] * __expf(cls[t] - clsS) : 0.f;
      G[t][sI] = (_Float16)g;
    }
  }
  __syncthreads();

  // ---- Phase 2+3: Y[t][p] = G*Xdt + exp(cls_t)*C*hstart^T ----
  const float* hst = hstart + ((size_t)bh * NC + c) * (HEADDIM * DSTATE);
  float Dh = Dskip[h];
  for (int q = wv; q < 16; q += 8) {
    int tt = q >> 2, tp = q & 3;
    int trow = tt * 16 + lr;                  // A row (t)
    int pcol = tp * 16 + lr;                  // B col (p)
    v8f acc = {};
    for (int k0 = 0; k0 < CHUNK; k0 += 32) {  // K = s
      v16h af, bf;
#pragma unroll
      for (int i = 0; i < 8; ++i) {
        af[i]     = G[trow][k0 + kbA + i];
        af[8 + i] = G[trow][k0 + 16 + kbA + i];
      }
#pragma unroll
      for (int j = 0; j < 16; ++j) {
        int s = k0 + kbB + (j & 7) + ((j >> 3) << 4);
        bf[j] = (_Float16)(XH[s][pcol] * dts[s]);
      }
      acc = __builtin_amdgcn_wmma_f32_16x16x32_f16(false, af, false, bf,
                                                   (short)0, acc, false, false);
    }
    float escl = __expf(cls[trow]);           // row-scale folded into A
    for (int k0 = 0; k0 < DSTATE; k0 += 32) { // K = n
      v16h af, bf;
#pragma unroll
      for (int i = 0; i < 8; ++i) {
        af[i]     = (_Float16)(CM[trow][k0 + kbA + i] * escl);
        af[8 + i] = (_Float16)(CM[trow][k0 + 16 + kbA + i] * escl);
      }
#pragma unroll
      for (int j = 0; j < 16; ++j)
        bf[j] = (_Float16)hst[(size_t)pcol * DSTATE + k0 + kbB + (j & 7) + ((j >> 3) << 4)];
      acc = __builtin_amdgcn_wmma_f32_16x16x32_f16(false, af, false, bf,
                                                   (short)0, acc, false, false);
    }
    int t0 = tt * 16 + hi * 8;
#pragma unroll
    for (int r = 0; r < 8; ++r) {
      int t = t0 + r;
      float xh = XH[t][pcol];
      ys[(rowbase + t) * DIN + h * HEADDIM + pcol] = acc[r] + Dh * xh;
    }
  }
}

// =====================================================================
// 8) Gate with silu(z) + RMSNorm (in place on ys). One row per block.
// =====================================================================
__global__ void gate_rmsnorm_kernel(float* __restrict__ ys,
                                    const float* __restrict__ zxbcdt,
                                    const float* __restrict__ nw) {
  int m = blockIdx.x;
  int tid = threadIdx.x;                      // 256, 2 elems each (DIN=512)
  __shared__ float red[256];
  size_t yb = (size_t)m * DIN;
  size_t zb = (size_t)m * DIP;
  float z0 = zxbcdt[zb + tid];
  float z1 = zxbcdt[zb + tid + 256];
  float y0 = ys[yb + tid] * (z0 / (1.f + __expf(-z0)));
  float y1 = ys[yb + tid + 256] * (z1 / (1.f + __expf(-z1)));
  red[tid] = y0 * y0 + y1 * y1;
  __syncthreads();
  for (int off = 128; off > 0; off >>= 1) {
    if (tid < off) red[tid] += red[tid + off];
    __syncthreads();
  }
  float scale = rsqrtf(red[0] / (float)DIN + EPS_);
  ys[yb + tid]       = y0 * scale * nw[tid];
  ys[yb + tid + 256] = y1 * scale * nw[tid + 256];
}

// =====================================================================
// 9) Output heads: track/peak dot products
// =====================================================================
__global__ void heads_kernel(const float* __restrict__ hbuf,
                             const float* __restrict__ tw, const float* __restrict__ tb,
                             const float* __restrict__ cw, const float* __restrict__ cb,
                             float* __restrict__ out) {
  int m = blockIdx.x;
  int tid = threadIdx.x;                      // 256 == D_
  __shared__ float r1[256];
  __shared__ float r2[256];
  float hv = hbuf[(size_t)m * D_ + tid];
  r1[tid] = hv * tw[tid];
  r2[tid] = hv * cw[tid];
  __syncthreads();
  for (int off = 128; off > 0; off >>= 1) {
    if (tid < off) { r1[tid] += r1[tid + off]; r2[tid] += r2[tid + off]; }
    __syncthreads();
  }
  if (tid == 0) {
    out[m]      = r1[0] + tb[0];
    out[BL + m] = r2[0] + cb[0];
  }
}

// =====================================================================
// Host-side orchestration
// =====================================================================
extern "C" void kernel_launch(void* const* d_in, const int* in_sizes, int n_in,
                              void* d_out, int out_size, void* d_ws, size_t ws_size,
                              hipStream_t stream) {
  (void)in_sizes; (void)n_in; (void)out_size; (void)ws_size;
  const int*   x       = (const int*)d_in[0];
  const float* emb     = (const float*)d_in[1];
  const float* pos     = (const float*)d_in[2];
  const float* W_in    = (const float*)d_in[3];
  const float* conv_w  = (const float*)d_in[4];
  const float* conv_b  = (const float*)d_in[5];
  const float* dt_bias = (const float*)d_in[6];
  const float* A_log   = (const float*)d_in[7];
  const float* D_skip  = (const float*)d_in[8];
  const float* norm_w  = (const float*)d_in[9];
  const float* W_out   = (const float*)d_in[10];
  const float* track_w = (const float*)d_in[11];
  const float* track_b = (const float*)d_in[12];
  const float* class_w = (const float*)d_in[13];
  const float* class_b = (const float*)d_in[14];
  float* out = (float*)d_out;

  char* ws = (char*)d_ws;
  size_t off = 0;
  auto alloc = [&](size_t bytes) -> void* {
    void* p = ws + off;
    off += (bytes + 255) & ~(size_t)255;
    return p;
  };
  float* hbuf   = (float*)alloc((size_t)BL * D_ * 4);          // 16 MB
  float* zxbcdt = (float*)alloc((size_t)BL * DIP * 4);         // 76 MB
  float* xBCc   = (float*)alloc((size_t)BL * CONVDIM * 4);     // 42 MB
  float* dtsp   = (float*)alloc((size_t)BL * NHEADS * 4);
  float* clsb   = (float*)alloc((size_t)BL * NHEADS * 4);
  float* Sc     = (float*)alloc((size_t)B_ * NHEADS * NC * HEADDIM * DSTATE * 4); // 32 MB
  float* ysb    = (float*)alloc((size_t)BL * DIN * 4);         // 32 MB
  float* hstart = Sc;  // state_scan is in-place safe (read-before-write)

  // Embedding
  embed_kernel<<<(BL * D_ + 255) / 256, 256, 0, stream>>>(x, emb, pos, hbuf);

  for (int i = 0; i < NLAYERS; ++i) {
    // In-projection: zxbcdt = h @ W_in^T  (M=16384, N=1160, K=256)
    {
      int ntn = (DIP + 15) / 16;
      int waves = (BL / 16) * ntn;
      int blocks = (waves + 7) / 8;           // 8 waves per 256-thread block
      gemm_nt_wmma<<<blocks, 256, 0, stream>>>(
          hbuf, W_in + (size_t)i * DIP * D_, zxbcdt, BL, DIP, D_);
    }
    // Depthwise causal conv + SiLU
    conv_silu_kernel<<<(BL * CONVDIM + 255) / 256, 256, 0, stream>>>(
        zxbcdt, conv_w + (size_t)i * CONVDIM * DCONV, conv_b + (size_t)i * CONVDIM, xBCc);
    // dt = softplus(raw + bias)
    dt_softplus_kernel<<<(BL * NHEADS + 255) / 256, 256, 0, stream>>>(
        zxbcdt, dt_bias + (size_t)i * NHEADS, dtsp);
    // per-chunk log-cumsum of dt*A
    clsum_kernel<<<(B_ * NHEADS * NC + 255) / 256, 256, 0, stream>>>(
        dtsp, A_log + (size_t)i * NHEADS, clsb);
    // chunk states (WMMA + TDM staging)
    chunk_state_wmma<<<B_ * NHEADS * NC, 256, 0, stream>>>(xBCc, dtsp, clsb, Sc);
    // inter-chunk recurrence (only serial part: 64 steps)
    state_scan_kernel<<<B_ * NHEADS, 256, 0, stream>>>(Sc, clsb, hstart);
    // intra-chunk outputs (WMMA x3 + TDM staging) + D skip
    y_chunk_wmma<<<B_ * NHEADS * NC, 256, 0, stream>>>(
        xBCc, dtsp, clsb, hstart, D_skip + (size_t)i * NHEADS, ysb);
    // gate + RMSNorm (in place)
    gate_rmsnorm_kernel<<<BL, 256, 0, stream>>>(ysb, zxbcdt, norm_w + (size_t)i * DIN);
    // Out-projection: h = yn @ W_out^T  (M=16384, N=256, K=512)
    {
      int ntn = (D_ + 15) / 16;
      int waves = (BL / 16) * ntn;
      int blocks = (waves + 7) / 8;
      gemm_nt_wmma<<<blocks, 256, 0, stream>>>(
          ysb, W_out + (size_t)i * D_ * DIN, hbuf, BL, D_, DIN);
    }
  }

  // Heads
  heads_kernel<<<BL, 256, 0, stream>>>(hbuf, track_w, track_b, class_w, class_b, out);
}
